// CausalFT_3135326126148
// MI455X (gfx1250) — matrix operand order repaired and checked by reading, not compiled, not run on hardware
//
#include <hip/hip_runtime.h>

// ---------------------------------------------------------------------------
// Problem constants (fixed by the reference)
// ---------------------------------------------------------------------------
#define QLEN  1024
#define MLEN  1024
#define KLEN  2048            // qlen + mlen
#define DMOD  1024
#define NBATCH 8
#define NCOL  (NBATCH * DMOD) // 8192 : flattened (b, d) == GEMM N dimension
#define FTLEN 1025            // klen - qlen + 1
#define KSTEPS 34             // band for a 64-row block: 34 K-blocks of 32

typedef __attribute__((ext_vector_type(16))) __bf16 v16bf;
typedef __attribute__((ext_vector_type(8)))  __bf16 v8bf;
typedef __attribute__((ext_vector_type(8)))  float  v8f;

// ---------------------------------------------------------------------------
// Kernel 1: build ft_mat [QLEN x KLEN] in bf16 from the closed form.
//   ft[i][j] = cos(2*pi * i*(j-i) / 1025) / sqrt(1025)   for i <= j <= i+1024
// i*(j-i) is reduced mod 1025 in integer arithmetic so the cos argument is
// always in [0, 2*pi) -- full fp32 accuracy before the bf16 round.
// ---------------------------------------------------------------------------
__global__ __launch_bounds__(256) void build_ftmat(__bf16* __restrict__ A) {
    int j = blockIdx.x * 256 + threadIdx.x;   // column 0..2047
    int i = blockIdx.y;                        // row    0..1023
    float v = 0.0f;
    if (j >= i && j <= i + (KLEN - QLEN)) {
        int r = (i * (j - i)) % FTLEN;                  // < 2^31, exact
        const float w = 6.28318530717958647692f / (float)FTLEN;
        v = __cosf((float)r * w) * 0.03123475237772121f; // 1/sqrt(1025)
    }
    A[(size_t)i * KLEN + j] = (__bf16)v;
}

// ---------------------------------------------------------------------------
// Kernel 2: build B^T = (concat(mems, dec_inp) + pos_emb)^T in bf16.
//   Bt[n][k], n = b*1024 + d (GEMM N), k = sequence position (GEMM K).
// LDS-tiled 32x32 transpose: global reads coalesced along d, writes
// coalesced along k.
// ---------------------------------------------------------------------------
__global__ __launch_bounds__(256) void build_catT(
    const float* __restrict__ dec, const float* __restrict__ pos,
    const float* __restrict__ mems, const int* __restrict__ addpos,
    __bf16* __restrict__ Bt)
{
    __shared__ float tile[32][33];
    const int k0 = blockIdx.x * 32;
    const int n0 = blockIdx.y * 32;
    const int tx = threadIdx.x, ty = threadIdx.y;     // block (32, 8)
    const bool ap = (*addpos) != 0;

    #pragma unroll
    for (int r = 0; r < 32; r += 8) {
        int k = k0 + ty + r;
        int n = n0 + tx;
        // mems/dec are [seq, b, d] => flat [seq][n] with n = b*1024+d
        const float* src = (k < MLEN) ? (mems + (size_t)k * NCOL)
                                      : (dec + (size_t)(k - MLEN) * NCOL);
        float v = src[n];
        if (ap) v += pos[(size_t)k * DMOD + (n & (DMOD - 1))];
        tile[ty + r][tx] = v;
    }
    __syncthreads();
    #pragma unroll
    for (int r = 0; r < 32; r += 8) {
        int n = n0 + ty + r;
        int k = k0 + tx;
        Bt[(size_t)n * KLEN + k] = (__bf16)tile[tx][ty + r];
    }
}

// ---------------------------------------------------------------------------
// Kernel 3: banded GEMM  ft = ft_mat @ cat  via v_wmma_f32_16x16x32_bf16,
// fused epilogue  out = dec_inp + ft / sqrt(2048).
//
// Block: 256 threads = 8 waves arranged 2(M) x 4(N); block tile 64 x 256.
// Each wave owns a 32x64 patch = 2x4 accumulator tiles (8 WMMAs / K-step
// from 2 A + 4 B fragment loads).
// Band: rows [BM0, BM0+64) only touch K in [BM0, BM0+1087] -> exactly 34
// K-blocks of 32 (BM0 is a multiple of 64, so already 32-aligned). The K
// loop is fully unrolled (constant trip count) so every fragment load is an
// immediate-offset global_load_b128 off a fixed base pointer (max offset
// 34*64 B = 2176 B, well inside the 24-bit field), and the scheduler can
// pipeline loads across WMMA groups. Whole working set (36 MB) is
// L2-resident (192 MB), so no software prefetch is needed.
//
// Fragment layouts as used here (wave32, half = lane/16, l16 = lane%16):
//   A (16x32 bf16): lane row = l16; elem e -> K = 16*half + e  (contiguous, as B)
//   B (32x16 bf16): lane col = l16; elem e -> K = 16*half + e  (contiguous!)
//   C/D (16x16 f32): elem r -> M = r + 8*half, N = l16
// Bt is pre-transposed so each B fragment is ONE contiguous 32B load.
// ---------------------------------------------------------------------------
__global__ __launch_bounds__(256) void gemm_ft(
    const __bf16* __restrict__ Abf, const __bf16* __restrict__ Bt,
    const float* __restrict__ dec, float* __restrict__ out)
{
    const int tid  = threadIdx.x;
    const int lane = tid & 31;
    const int half = lane >> 4;
    const int l16  = lane & 15;
    const int wave = tid >> 5;
    const int wr   = wave >> 2;       // 0..1  (M)
    const int wc   = wave & 3;        // 0..3  (N)

    const int BM0 = blockIdx.y * 64;
    const int BN0 = blockIdx.x * 256;
    const int wM  = BM0 + wr * 32;
    const int wN  = BN0 + wc * 64;

    // Fixed base pointers at k = BM0 (start of this block's band).
    const __bf16* pa[2];
    #pragma unroll
    for (int mi = 0; mi < 2; ++mi)
        pa[mi] = Abf + (size_t)(wM + mi * 16 + l16) * KLEN + BM0 + 16 * half;
    const __bf16* pb[4];
    #pragma unroll
    for (int ni = 0; ni < 4; ++ni)
        pb[ni] = Bt + (size_t)(wN + ni * 16 + l16) * KLEN + BM0 + 16 * half;

    v8f acc[2][4] = {};

    #pragma unroll
    for (int t = 0; t < KSTEPS; ++t) {
        const int koff = t * 32;      // elements; compile-time per unroll step
        v16bf a[2];
        #pragma unroll
        for (int mi = 0; mi < 2; ++mi) {
            union { v16bf v; v8bf h[2]; } ua;
            ua.h[0] = *(const v8bf*)(pa[mi] + koff);       // K = 16*half+0..7
            ua.h[1] = *(const v8bf*)(pa[mi] + koff + 8);   // K = 16*half+8..15
            a[mi] = ua.v;
        }
        v16bf b[4];
        #pragma unroll
        for (int ni = 0; ni < 4; ++ni)
            b[ni] = *(const v16bf*)(pb[ni] + koff);
        #pragma unroll
        for (int mi = 0; mi < 2; ++mi)
            #pragma unroll
            for (int ni = 0; ni < 4; ++ni)
                acc[mi][ni] = __builtin_amdgcn_wmma_f32_16x16x32_bf16(
                    false, a[mi], false, b[ni], (short)0, acc[mi][ni],
                    false, false);
    }

    // ---- epilogue: out[m][n] = dec[m][n] + acc / sqrt(2048) ----
    const float rscale = 0.0220970869120796101f;   // 1/sqrt(2048)
    #pragma unroll
    for (int mi = 0; mi < 2; ++mi) {
        #pragma unroll
        for (int ni = 0; ni < 4; ++ni) {
            const int n = wN + ni * 16 + l16;
            #pragma unroll
            for (int r = 0; r < 8; ++r) {
                const int m = wM + mi * 16 + 8 * half + r;
                const size_t idx = (size_t)m * NCOL + n;
                out[idx] = dec[idx] + acc[mi][ni][r] * rscale;
            }
        }
    }
}

// ---------------------------------------------------------------------------
// Kernel 4: in-place LayerNorm over the last dim (1024) of out.
// One 256-thread block per (m, b) row; wave32 shuffle + LDS reduction.
// Safe in place: each thread rewrites exactly the elements it read.
// ---------------------------------------------------------------------------
__global__ __launch_bounds__(256) void layernorm_inplace(
    float* __restrict__ out, const float* __restrict__ gamma,
    const float* __restrict__ beta)
{
    float* x = out + (size_t)blockIdx.x * DMOD;
    const int tid = threadIdx.x;

    float4 v = ((const float4*)x)[tid];
    float s  = v.x + v.y + v.z + v.w;
    float s2 = v.x * v.x + v.y * v.y + v.z * v.z + v.w * v.w;

    #pragma unroll
    for (int off = 16; off > 0; off >>= 1) {
        s  += __shfl_down(s,  off, 32);
        s2 += __shfl_down(s2, off, 32);
    }
    __shared__ float ws[8], ws2[8];
    const int w = tid >> 5, l = tid & 31;
    if (l == 0) { ws[w] = s; ws2[w] = s2; }
    __syncthreads();
    if (w == 0) {
        float a  = (l < 8) ? ws[l]  : 0.0f;
        float a2 = (l < 8) ? ws2[l] : 0.0f;
        #pragma unroll
        for (int off = 4; off > 0; off >>= 1) {
            a  += __shfl_down(a,  off, 32);
            a2 += __shfl_down(a2, off, 32);
        }
        if (l == 0) { ws[0] = a; ws2[0] = a2; }
    }
    __syncthreads();
    const float mu  = ws[0] * (1.0f / DMOD);
    const float var = ws2[0] * (1.0f / DMOD) - mu * mu;
    const float inv = rsqrtf(var + 1e-5f);

    float4 g = ((const float4*)gamma)[tid];
    float4 bb = ((const float4*)beta)[tid];
    float4 r;
    r.x = g.x * (v.x - mu) * inv + bb.x;
    r.y = g.y * (v.y - mu) * inv + bb.y;
    r.z = g.z * (v.z - mu) * inv + bb.z;
    r.w = g.w * (v.w - mu) * inv + bb.w;
    ((float4*)x)[tid] = r;
}

// ---------------------------------------------------------------------------
// Launch. Inputs (setup_inputs order):
//   0 dec_inp [1024,8,1024] f32   1 pos_emb [2048,1,1024] f32
//   2 mems    [1024,8,1024] f32   3 gamma [1024] f32   4 beta [1024] f32
//   5 add_position (int scalar)
// Workspace: ft_mat bf16 (4 MB) + catT bf16 (32 MB) = 36 MB.
// ---------------------------------------------------------------------------
extern "C" void kernel_launch(void* const* d_in, const int* in_sizes, int n_in,
                              void* d_out, int out_size, void* d_ws, size_t ws_size,
                              hipStream_t stream) {
    const float* dec   = (const float*)d_in[0];
    const float* pos   = (const float*)d_in[1];
    const float* mems  = (const float*)d_in[2];
    const float* gamma = (const float*)d_in[3];
    const float* beta  = (const float*)d_in[4];
    const int*   addp  = (const int*)  d_in[5];
    float* out = (float*)d_out;

    __bf16* Abf = (__bf16*)d_ws;
    __bf16* Bt  = (__bf16*)((char*)d_ws + (size_t)QLEN * KLEN * sizeof(__bf16));

    build_ftmat<<<dim3(KLEN / 256, QLEN), 256, 0, stream>>>(Abf);
    build_catT<<<dim3(KLEN / 32, NCOL / 32), dim3(32, 8), 0, stream>>>(
        dec, pos, mems, addp, Bt);
    gemm_ft<<<dim3(NCOL / 256, QLEN / 64), 256, 0, stream>>>(Abf, Bt, dec, out);
    layernorm_inplace<<<QLEN * NBATCH, 256, 0, stream>>>(out, gamma, beta);
}
